// EGNN_5892695130664
// MI455X (gfx1250) — compile-verified, matrix operand order checked by reading, not run on hardware
//
#include <hip/hip_runtime.h>
#include <hip/hip_bf16.h>

#define HIDN   128
#define NNODES 25
#define BGR    512
#define NTOT   (BGR*NNODES)      // 12800
#define EPG    (NNODES*(NNODES-1)) // 600
#define ETOT   (BGR*EPG)         // 307200
#define KEPAD  288               // 267 padded to 9*32
#define PADE   29
#define PADD   30
#define FBIG   3.0e38f

typedef __attribute__((ext_vector_type(16))) _Float16 v16h;
typedef __attribute__((ext_vector_type(8)))  _Float16 v8h;
typedef __attribute__((ext_vector_type(8)))  float    v8f;

__device__ __forceinline__ float silu_f(float v) { return v / (1.0f + __expf(-v)); }

// A fragment (16x32 f16) from row-major f16 LDS tile, per ISA 7.12.2 layout:
// lane L holds row M=L%16; halves 0..7 = K kb+lo..kb+lo+7, halves 8..15 = K kb+16+lo..
__device__ __forceinline__ v16h load_a_frag(const _Float16* s, int strideH,
                                            int rowBase, int kBase, int lane) {
  int m  = rowBase + (lane & 15);
  int lo = (lane & 16) ? 8 : 0;
  const _Float16* p = s + m * strideH + kBase + lo;
  v8h x = *(const v8h*)(p);
  v8h y = *(const v8h*)(p + 16);
  v16h a;
#pragma unroll
  for (int i = 0; i < 8; ++i) { a[i] = x[i]; a[i + 8] = y[i]; }
  return a;
}

// B fragments are pre-shuffled into fragment order: one contiguous 32B load.
__device__ __forceinline__ v16h load_b_frag(const _Float16* pack, int kt, int nt, int lane) {
  return *(const v16h*)(pack + (size_t)(((kt * 8 + nt) * 32 + lane) * 16));
}

__device__ __forceinline__ v8f wmma16(v16h a, v16h b, v8f c) {
  return __builtin_amdgcn_wmma_f32_16x16x32_f16(false, a, false, b, (short)0, c, false, false);
}

// Fully unrolled ascending bitonic sort of 32 register-resident floats.
__device__ __forceinline__ void sort32(float* a) {
#pragma unroll
  for (int k = 2; k <= 32; k <<= 1) {
#pragma unroll
    for (int j = k >> 1; j > 0; j >>= 1) {
#pragma unroll
      for (int i = 0; i < 32; ++i) {
        int ixj = i ^ j;
        if (ixj > i) {
          float x = a[i], y = a[ixj];
          float lo2 = fminf(x, y), hi2 = fmaxf(x, y);
          bool up = ((i & k) == 0);
          a[i]   = up ? lo2 : hi2;
          a[ixj] = up ? hi2 : lo2;
        }
      }
    }
  }
}

// ---------------- prep kernels ----------------

// Pack a (din x 128) f32 weight matrix into WMMA-B fragment order (f16), K padded to ktiles*32.
__global__ void k_pack(const float* __restrict__ W, int din, int ktiles, _Float16* __restrict__ dst) {
  int t = blockIdx.x * 256 + threadIdx.x;
  int total = ktiles * 8 * 32 * 16;
  if (t >= total) return;
  int hh   = t & 15;
  int lane = (t >> 4) & 31;
  int nt   = (t >> 9) & 7;
  int kt   = t >> 12;
  int lo   = (lane & 16) ? 8 : 0;
  int kk   = (hh < 8) ? hh : hh + 8;
  int K    = kt * 32 + lo + kk;
  int n    = nt * 16 + (lane & 15);
  float v  = (K < din) ? W[K * 128 + n] : 0.0f;
  dst[t] = (_Float16)v;
}

__global__ void k_center(const float* __restrict__ x, float* __restrict__ cen) {
  int g = blockIdx.x * blockDim.x + threadIdx.x;
  if (g >= BGR) return;
  float mx = 0, my = 0, mz = 0;
  for (int n = 0; n < NNODES; ++n) {
    mx += x[(g * NNODES + n) * 3 + 0];
    my += x[(g * NNODES + n) * 3 + 1];
    mz += x[(g * NNODES + n) * 3 + 2];
  }
  mx /= NNODES; my /= NNODES; mz /= NNODES;
  for (int n = 0; n < NNODES; ++n) {
    cen[(g * NNODES + n) * 3 + 0] = x[(g * NNODES + n) * 3 + 0] - mx;
    cen[(g * NNODES + n) * 3 + 1] = x[(g * NNODES + n) * 3 + 1] - my;
    cen[(g * NNODES + n) * 3 + 2] = x[(g * NNODES + n) * 3 + 2] - mz;
  }
}

__global__ void k_embed_h(const float* __restrict__ h0, const float* __restrict__ eW,
                          const float* __restrict__ eb, float* __restrict__ h) {
  int t = blockIdx.x * 256 + threadIdx.x;   // exact: NTOT*128
  int n = t >> 7, c = t & 127;
  float a = eb[c];
#pragma unroll
  for (int d = 0; d < 11; ++d) a += h0[n * 11 + d] * eW[d * 128 + c];
  h[t] = a;
}

__global__ void k_radial(const float* __restrict__ x, const int* __restrict__ rowI,
                         const int* __restrict__ colI, float* __restrict__ rad) {
  int e = blockIdx.x * 256 + threadIdx.x;   // exact: ETOT
  int r = rowI[e], c = colI[e];
  float dx = x[r * 3 + 0] - x[c * 3 + 0];
  float dy = x[r * 3 + 1] - x[c * 3 + 1];
  float dz = x[r * 3 + 2] - x[c * 3 + 2];
  rad[e] = dx * dx + dy * dy + dz * dz;
}

// Sorted embedding per (edge, channel): rotation-invariant projections, mask, sort29, dot v.
__global__ __launch_bounds__(256) void k_embsort(const int* __restrict__ rowI,
    const int* __restrict__ colI, const float* __restrict__ cen,
    const float* __restrict__ W, const float* __restrict__ vv, float* __restrict__ embe) {
  int t = blockIdx.x * 256 + threadIdx.x;   // exact: ETOT*8
  int e = t >> 3, o = t & 7;
  int re = rowI[e], ce = colI[e];
  int g  = e / EPG;
  float f1x = cen[re * 3 + 0], f1y = cen[re * 3 + 1], f1z = cen[re * 3 + 2];
  float f2x = cen[ce * 3 + 0], f2y = cen[ce * 3 + 1], f2z = cen[ce * 3 + 2];
  float f3x = f1y * f2z - f1z * f2y;
  float f3y = f1z * f2x - f1x * f2z;
  float f3z = f1x * f2y - f1y * f2x;
  float w0 = W[o * 3 + 0], w1 = W[o * 3 + 1], w2 = W[o * 3 + 2];
  const float* cg = cen + g * NNODES * 3;
  float val[32];
#pragma unroll
  for (int n = 0; n < 32; ++n) val[n] = (n < PADE) ? 0.0f : FBIG;
#pragma unroll
  for (int n = 0; n < NNODES; ++n) {
    float cx = cg[n * 3 + 0], cy = cg[n * 3 + 1], cz = cg[n * 3 + 2];
    float p0 = f1x * cx + f1y * cy + f1z * cz;
    float p1 = f2x * cx + f2y * cy + f2z * cz;
    float p2 = f3x * cx + f3y * cy + f3z * cz;
    float m  = ((fabsf(p0) + fabsf(p1) + fabsf(p2)) > 0.0f) ? 1.0f : 0.0f;
    val[n] = (w0 * p0 + w1 * p1 + w2 * p2) * m;
  }
  sort32(val);
  float s = 0.0f;
#pragma unroll
  for (int n = 0; n < PADE; ++n) s += val[n] * vv[n];
  embe[e * 8 + o] = s;
}

// ---------------- edge MLP + segment reduce (one block per node, 4 waves) ----------------
__global__ __launch_bounds__(128) void k_edge(const float* __restrict__ h,
    const float* __restrict__ rad, const float* __restrict__ embe,
    const float* __restrict__ eattr, const float* __restrict__ emask,
    const int* __restrict__ colI,
    const _Float16* __restrict__ e1p, const float* __restrict__ e1b,
    const _Float16* __restrict__ e2p, const float* __restrict__ e2b,
    float* __restrict__ agg) {
  __shared__ _Float16 sA[32 * KEPAD];
  __shared__ _Float16 sM[32 * 128];
  __shared__ float    sR[32 * 128];
  int tid   = threadIdx.x;
  int node  = blockIdx.x;
  int g     = node / NNODES, i = node % NNODES;
  int ebase = g * EPG + i * 24;

  for (int idx = tid; idx < 32 * KEPAD; idx += 128) {
    int r = idx / KEPAD, k = idx - r * KEPAD;
    float v = 0.0f;
    if (r < 24) {
      int e = ebase + r;
      if (k < 128)        v = h[node * 128 + k];
      else if (k < 256)   v = h[colI[e] * 128 + (k - 128)];
      else if (k == 256)  v = rad[e];
      else if (k < 265)   v = embe[e * 8 + (k - 257)];
      else if (k < 267)   v = eattr[e * 2 + (k - 265)];
    }
    sA[idx] = (_Float16)v;
  }
  __syncthreads();

  int lane = tid & 31, wv = tid >> 5;
  v8f z8 = {0, 0, 0, 0, 0, 0, 0, 0};
  v8f acc00 = z8, acc01 = z8, acc10 = z8, acc11 = z8;
#pragma unroll
  for (int kt = 0; kt < 9; ++kt) {
    v16h a0 = load_a_frag(sA, KEPAD, 0,  kt * 32, lane);
    v16h a1 = load_a_frag(sA, KEPAD, 16, kt * 32, lane);
    v16h b0 = load_b_frag(e1p, kt, wv * 2 + 0, lane);
    v16h b1 = load_b_frag(e1p, kt, wv * 2 + 1, lane);
    acc00 = wmma16(a0, b0, acc00);
    acc01 = wmma16(a0, b1, acc01);
    acc10 = wmma16(a1, b0, acc10);
    acc11 = wmma16(a1, b1, acc11);
  }
  int cb0 = (wv * 2) * 16 + (lane & 15), cb1 = cb0 + 16;
  int ro  = (lane & 16) ? 8 : 0;
  float bb0 = e1b[cb0], bb1 = e1b[cb1];
#pragma unroll
  for (int r = 0; r < 8; ++r) {
    int r0 = r + ro, r1 = 16 + r + ro;
    sM[r0 * 128 + cb0] = (_Float16)silu_f(acc00[r] + bb0);
    sM[r0 * 128 + cb1] = (_Float16)silu_f(acc01[r] + bb1);
    sM[r1 * 128 + cb0] = (_Float16)silu_f(acc10[r] + bb0);
    sM[r1 * 128 + cb1] = (_Float16)silu_f(acc11[r] + bb1);
  }
  __syncthreads();

  acc00 = z8; acc01 = z8; acc10 = z8; acc11 = z8;
#pragma unroll
  for (int kt = 0; kt < 4; ++kt) {
    v16h a0 = load_a_frag(sM, 128, 0,  kt * 32, lane);
    v16h a1 = load_a_frag(sM, 128, 16, kt * 32, lane);
    v16h b0 = load_b_frag(e2p, kt, wv * 2 + 0, lane);
    v16h b1 = load_b_frag(e2p, kt, wv * 2 + 1, lane);
    acc00 = wmma16(a0, b0, acc00);
    acc01 = wmma16(a0, b1, acc01);
    acc10 = wmma16(a1, b0, acc10);
    acc11 = wmma16(a1, b1, acc11);
  }
  float c20 = e2b[cb0], c21 = e2b[cb1];
#pragma unroll
  for (int r = 0; r < 8; ++r) {
    int r0 = r + ro, r1 = 16 + r + ro;
    float m0 = emask[ebase + r0];                      // rows 0..15 always valid
    float m1 = (r1 < 24) ? emask[ebase + r1] : 0.0f;   // rows 24..31 are padding
    sR[r0 * 128 + cb0] = silu_f(acc00[r] + c20) * m0;
    sR[r0 * 128 + cb1] = silu_f(acc01[r] + c21) * m0;
    sR[r1 * 128 + cb0] = silu_f(acc10[r] + c20) * m1;
    sR[r1 * 128 + cb1] = silu_f(acc11[r] + c21) * m1;
  }
  __syncthreads();
  float s = 0.0f;
#pragma unroll
  for (int r = 0; r < 24; ++r) s += sR[r * 128 + tid];
  agg[node * 128 + tid] = s;   // segment_sum: 24 consecutive edges per node, no atomics
}

// ---------------- node MLP (MODE 0) / decoder MLP (MODE 1), 32 nodes per block ----------------
template <int MODE>
__global__ __launch_bounds__(128) void k_nodemlp(const float* __restrict__ h,
    const float* __restrict__ agg, const float* __restrict__ h0,
    const float* __restrict__ nmask,
    const _Float16* __restrict__ w1p, const float* __restrict__ b1v,
    const _Float16* __restrict__ w2p, const float* __restrict__ b2v,
    float* __restrict__ outp) {
  __shared__ _Float16 sA[32 * KEPAD];
  __shared__ _Float16 sM[32 * 128];
  int tid = threadIdx.x;
  int nb  = blockIdx.x * 32;
  if (MODE == 0) {
    for (int idx = tid; idx < 32 * KEPAD; idx += 128) {
      int r = idx / KEPAD, k = idx - r * KEPAD;
      int node = nb + r;
      float v = 0.0f;
      if (k < 128)      v = h[node * 128 + k];
      else if (k < 256) v = agg[node * 128 + (k - 128)];
      else if (k < 267) v = h0[node * 11 + (k - 256)];
      sA[idx] = (_Float16)v;
    }
  } else {
    for (int idx = tid; idx < 32 * 128; idx += 128) {
      int r = idx >> 7, k = idx & 127;
      sA[r * KEPAD + k] = (_Float16)h[(nb + r) * 128 + k];
    }
  }
  __syncthreads();

  int lane = tid & 31, wv = tid >> 5;
  v8f z8 = {0, 0, 0, 0, 0, 0, 0, 0};
  v8f acc00 = z8, acc01 = z8, acc10 = z8, acc11 = z8;
  const int KT1 = (MODE == 0) ? 9 : 4;
#pragma unroll
  for (int kt = 0; kt < KT1; ++kt) {
    v16h a0 = load_a_frag(sA, KEPAD, 0,  kt * 32, lane);
    v16h a1 = load_a_frag(sA, KEPAD, 16, kt * 32, lane);
    v16h b0 = load_b_frag(w1p, kt, wv * 2 + 0, lane);
    v16h b1 = load_b_frag(w1p, kt, wv * 2 + 1, lane);
    acc00 = wmma16(a0, b0, acc00);
    acc01 = wmma16(a0, b1, acc01);
    acc10 = wmma16(a1, b0, acc10);
    acc11 = wmma16(a1, b1, acc11);
  }
  int cb0 = (wv * 2) * 16 + (lane & 15), cb1 = cb0 + 16;
  int ro  = (lane & 16) ? 8 : 0;
  float bb0 = b1v[cb0], bb1 = b1v[cb1];
#pragma unroll
  for (int r = 0; r < 8; ++r) {
    int r0 = r + ro, r1 = 16 + r + ro;
    sM[r0 * 128 + cb0] = (_Float16)silu_f(acc00[r] + bb0);
    sM[r0 * 128 + cb1] = (_Float16)silu_f(acc01[r] + bb1);
    sM[r1 * 128 + cb0] = (_Float16)silu_f(acc10[r] + bb0);
    sM[r1 * 128 + cb1] = (_Float16)silu_f(acc11[r] + bb1);
  }
  __syncthreads();
  acc00 = z8; acc01 = z8; acc10 = z8; acc11 = z8;
#pragma unroll
  for (int kt = 0; kt < 4; ++kt) {
    v16h a0 = load_a_frag(sM, 128, 0,  kt * 32, lane);
    v16h a1 = load_a_frag(sM, 128, 16, kt * 32, lane);
    v16h b0 = load_b_frag(w2p, kt, wv * 2 + 0, lane);
    v16h b1 = load_b_frag(w2p, kt, wv * 2 + 1, lane);
    acc00 = wmma16(a0, b0, acc00);
    acc01 = wmma16(a0, b1, acc01);
    acc10 = wmma16(a1, b0, acc10);
    acc11 = wmma16(a1, b1, acc11);
  }
  float d0 = b2v[cb0], d1 = b2v[cb1];
#pragma unroll
  for (int r = 0; r < 8; ++r) {
    int r0 = r + ro, r1 = 16 + r + ro;
    int n0 = nb + r0, n1 = nb + r1;
    if (MODE == 0) {                       // h = h + mlp(n_in)  (no silu on second layer)
      outp[n0 * 128 + cb0] += acc00[r] + d0;
      outp[n0 * 128 + cb1] += acc01[r] + d1;
      outp[n1 * 128 + cb0] += acc10[r] + d0;
      outp[n1 * 128 + cb1] += acc11[r] + d1;
    } else {                               // h2 = (silu(h@dec1+b)@dec2+b) * node_mask
      float q0 = nmask[n0], q1 = nmask[n1];
      outp[n0 * 128 + cb0] = (acc00[r] + d0) * q0;
      outp[n0 * 128 + cb1] = (acc01[r] + d1) * q0;
      outp[n1 * 128 + cb0] = (acc10[r] + d0) * q1;
      outp[n1 * 128 + cb1] = (acc11[r] + d1) * q1;
    }
  }
}

// ---------------- decoder sorted embedding: one block per graph, thread per output channel ----------------
__global__ __launch_bounds__(128) void k_decsort(const float* __restrict__ h2,
    const float* __restrict__ dW, const float* __restrict__ dv, float* __restrict__ hdec) {
  __shared__ float sh[NNODES * 128];
  __shared__ float sm[NNODES];
  int g = blockIdx.x, tid = threadIdx.x;
  for (int idx = tid; idx < NNODES * 128; idx += 128) sh[idx] = h2[g * NNODES * 128 + idx];
  __syncthreads();
  if (tid < NNODES) {
    float s = 0.0f;
    for (int d = 0; d < 128; ++d) s += fabsf(sh[tid * 128 + d]);
    sm[tid] = (s > 0.0f) ? 1.0f : 0.0f;
  }
  __syncthreads();
  float val[32];
#pragma unroll
  for (int n = 0; n < 32; ++n) val[n] = (n < PADD) ? 0.0f : FBIG;
  for (int d = 0; d < 128; ++d) {
    float wd = dW[tid * 128 + d];
#pragma unroll
    for (int n = 0; n < NNODES; ++n) val[n] += wd * sh[n * 128 + d];
  }
#pragma unroll
  for (int n = 0; n < NNODES; ++n) val[n] *= sm[n];
  sort32(val);
  float s = 0.0f;
#pragma unroll
  for (int n = 0; n < PADD; ++n) s += val[n] * dv[n];
  hdec[g * 128 + tid] = s;
}

__global__ __launch_bounds__(128) void k_head(const float* __restrict__ hdec,
    const float* __restrict__ g1W, const float* __restrict__ g1b,
    const float* __restrict__ g2W, const float* __restrict__ g2b, float* __restrict__ out) {
  __shared__ float sd[128];
  __shared__ float sr[128];
  int g = blockIdx.x, c = threadIdx.x;
  sd[c] = hdec[g * 128 + c];
  __syncthreads();
  float z = g1b[c];
  for (int d = 0; d < 128; ++d) z += sd[d] * g1W[d * 128 + c];
  sr[c] = silu_f(z) * g2W[c];
  __syncthreads();
  for (int st = 64; st > 0; st >>= 1) {
    if (c < st) sr[c] += sr[c + st];
    __syncthreads();
  }
  if (c == 0) out[g] = sr[0] + g2b[0];
}

// ---------------- host ----------------
extern "C" void kernel_launch(void* const* d_in, const int* in_sizes, int n_in,
                              void* d_out, int out_size, void* d_ws, size_t ws_size,
                              hipStream_t stream) {
  (void)out_size; (void)ws_size;
  const float* h0    = (const float*)d_in[0];
  const float* x     = (const float*)d_in[1];
  const float* eattr = (const float*)d_in[2];
  const float* nmask = (const float*)d_in[3];
  const float* emask = (const float*)d_in[4];
  const int*   rowI  = (const int*)d_in[n_in - 3];
  const int*   colI  = (const int*)d_in[n_in - 2];

  // Params pytree flattening: sorted-key (jax.tree_leaves) vs insertion order.
  const float *dec1W, *dec1b, *dec2W, *dec2b, *decW, *decv, *embW, *embb, *g1W, *g1b, *g2W, *g2b;
  struct Lp { const float *e1W, *e1b, *e2W, *e2b, *emW, *emv, *n1W, *n1b, *n2W, *n2b; } L[4];
  auto f = [&](int i) { return (const float*)d_in[i]; };
  bool sortedKeys = (in_sizes[5] == 128 * 128);   // dec1.W first if sorted; emb.W (1408) if insertion
  if (sortedKeys) {
    dec1W = f(5);  dec1b = f(6);  dec2W = f(7);  dec2b = f(8);
    decW  = f(9);  decv  = f(10); embW  = f(11); embb  = f(12);
    g1W   = f(13); g1b   = f(14); g2W   = f(15); g2b   = f(16);
    for (int l = 0; l < 4; ++l) {
      int b = 17 + 10 * l;
      L[l] = { f(b), f(b+1), f(b+2), f(b+3), f(b+4), f(b+5), f(b+6), f(b+7), f(b+8), f(b+9) };
    }
  } else {
    embW = f(5); embb = f(6);
    for (int l = 0; l < 4; ++l) {
      int b = 7 + 10 * l;   // e1W,e1b,e2W,e2b,n1W,n1b,n2W,n2b,embW,embv
      L[l] = { f(b), f(b+1), f(b+2), f(b+3), f(b+8), f(b+9), f(b+4), f(b+5), f(b+6), f(b+7) };
    }
    dec1W = f(47); dec1b = f(48); dec2W = f(49); dec2b = f(50);
    decW  = f(51); decv  = f(52); g1W   = f(53); g1b   = f(54);
    g2W   = f(55); g2b   = f(56);
  }

  // Workspace carve-up (~25 MB)
  size_t off = 0;
  auto take = [&](size_t bytes) -> void* {
    void* p = (char*)d_ws + off;
    off = (off + bytes + 255) & ~(size_t)255;
    return p;
  };
  float* hbuf   = (float*)take((size_t)NTOT * 128 * 4);
  float* aggbuf = (float*)take((size_t)NTOT * 128 * 4);   // reused as h2 after layers
  float* cenbuf = (float*)take((size_t)NTOT * 3 * 4);
  float* radbuf = (float*)take((size_t)ETOT * 4);
  float* embbuf = (float*)take((size_t)ETOT * 8 * 4);
  float* hdecb  = (float*)take((size_t)BGR * 128 * 4);
  _Float16* pk  = (_Float16*)take((size_t)(4 * 106496 + 2 * 16384) * 2);

  const int E1H = 9 * 8 * 32 * 16;   // 36864 halves
  const int SQH = 4 * 8 * 32 * 16;   // 16384 halves
  _Float16 *e1p[4], *e2p[4], *n1p[4], *n2p[4];
  for (int l = 0; l < 4; ++l) {
    _Float16* base = pk + (size_t)l * (E1H + SQH + E1H + SQH);
    e1p[l] = base;
    e2p[l] = base + E1H;
    n1p[l] = base + E1H + SQH;
    n2p[l] = base + E1H + SQH + E1H;
  }
  _Float16* d1p = pk + (size_t)4 * (E1H + SQH + E1H + SQH);
  _Float16* d2p = d1p + SQH;

  // Pack weights into WMMA-B fragment order
  for (int l = 0; l < 4; ++l) {
    k_pack<<<144, 256, 0, stream>>>(L[l].e1W, 267, 9, e1p[l]);
    k_pack<<< 64, 256, 0, stream>>>(L[l].e2W, 128, 4, e2p[l]);
    k_pack<<<144, 256, 0, stream>>>(L[l].n1W, 267, 9, n1p[l]);
    k_pack<<< 64, 256, 0, stream>>>(L[l].n2W, 128, 4, n2p[l]);
  }
  k_pack<<<64, 256, 0, stream>>>(dec1W, 128, 4, d1p);
  k_pack<<<64, 256, 0, stream>>>(dec2W, 128, 4, d2p);

  k_center <<<2, 256, 0, stream>>>(x, cenbuf);
  k_embed_h<<<(NTOT * 128) / 256, 256, 0, stream>>>(h0, embW, embb, hbuf);
  k_radial <<<ETOT / 256, 256, 0, stream>>>(x, rowI, colI, radbuf);

  for (int l = 0; l < 4; ++l) {
    k_embsort<<<(ETOT * 8) / 256, 256, 0, stream>>>(rowI, colI, cenbuf, L[l].emW, L[l].emv, embbuf);
    k_edge<<<NTOT, 128, 0, stream>>>(hbuf, radbuf, embbuf, eattr, emask, colI,
                                     e1p[l], L[l].e1b, e2p[l], L[l].e2b, aggbuf);
    k_nodemlp<0><<<NTOT / 32, 128, 0, stream>>>(hbuf, aggbuf, h0, nmask,
                                                n1p[l], L[l].n1b, n2p[l], L[l].n2b, hbuf);
  }

  k_nodemlp<1><<<NTOT / 32, 128, 0, stream>>>(hbuf, aggbuf, h0, nmask,
                                              d1p, dec1b, d2p, dec2b, aggbuf);
  k_decsort<<<BGR, 128, 0, stream>>>(aggbuf, decW, decv, hdecb);
  k_head  <<<BGR, 128, 0, stream>>>(hdecb, g1W, g1b, g2W, g2b, (float*)d_out);
}
